// RPN_24395414241610
// MI455X (gfx1250) — matrix operand order repaired and checked by reading, not compiled, not run on hardware
//
#include <hip/hip_runtime.h>
#include <math.h>

// ---------------- types for CDNA5 WMMA ----------------
typedef __bf16 bf16;
typedef __attribute__((ext_vector_type(16))) __bf16 v16bf;
typedef __attribute__((ext_vector_type(8)))  float  v8f;

// ---------------- problem constants ----------------
#define TOTALP   87296        // sum of H*W over 5 levels
#define TOTALA   261888       // 3 * TOTALP anchors
#define KPRE     6000
#define KPOST    1000
#define CAP      8192         // power-of-2 >= KPRE (bitonic sort size)
#define NWORDS   188          // ceil(6000/32)

// ---------------- workspace layout (bytes) ----------------
#define OFF_FT      ((size_t)0)             // bf16 feats, position-major: 87296*256*2 = 44,695,552
#define OFF_WPACK   ((size_t)44695552)      // bf16 packed conv weights: 1,179,648
#define OFF_SCORES  ((size_t)45875200)      // f32 scores: 261888*4
#define OFF_DELTAS  ((size_t)46922752)      // f32 deltas: 261888*4*4
#define OFF_SELK    ((size_t)51112960)      // u64 sort keys: 8192*8
#define OFF_ORDER   ((size_t)51178496)      // u32 order: 6000*4 (padded)
#define OFF_BOXES   ((size_t)51202560)      // f32 boxes: 6000*4*4 (padded)
#define OFF_MASK    ((size_t)51298816)      // u32 suppression mask: 6000*188*4
#define OFF_KEEP    ((size_t)55810816)      // u32 keep: 6000*4 (padded)
#define OFF_MISC    ((size_t)55834880)      // [0]=threshold key, [1]=gather counter
#define WS_NEEDED   ((size_t)55835136)

// =====================================================================
// 1) Pack 3x3 conv weights into this file's contiguous-k WMMA A-operand lane layout.
//    Entry t = ((ot*9+tap)*8+kk)*32+lane holds 16 bf16 for that lane:
//    lanes 0-15: K = {0..15}; lanes 16-31: K = {16..31}.
// =====================================================================
__global__ void k_pack_w(const float* __restrict__ w0, bf16* __restrict__ wpack) {
    int t = blockIdx.x * blockDim.x + threadIdx.x;
    if (t >= 16 * 9 * 8 * 32) return;
    int lane = t & 31;
    int kk   = (t >> 5) & 7;
    int tap  = (t >> 8) % 9;
    int ot   = t / 2304;
    int m    = lane & 15;
    int half = lane >> 4;
    int ky = tap / 3, kx = tap % 3;
    int oc = ot * 16 + m;
    bf16* dst = wpack + (size_t)t * 16;
    #pragma unroll
    for (int e = 0; e < 16; ++e) {
        int kin = half * 16 + e;
        int ci = kk * 32 + kin;
        dst[e] = (bf16)w0[((oc * 256 + ci) * 3 + ky) * 3 + kx];
    }
}

// =====================================================================
// 2) Transpose NCHW f32 features of all levels -> position-major bf16:
//    ft[(posBase+p)*256 + c]  (reads coalesced along p).
// =====================================================================
__global__ void k_transpose(const float* __restrict__ f0, const float* __restrict__ f1,
                            const float* __restrict__ f2, const float* __restrict__ f3,
                            const float* __restrict__ f4, bf16* __restrict__ ft) {
    const int posBase[5] = {0, 65536, 81920, 86016, 87040};
    const int Wl[5]      = {256, 128, 64, 32, 16};
    long long idx = (long long)blockIdx.x * blockDim.x + threadIdx.x;
    if (idx >= (long long)TOTALP * 256) return;
    int c = (int)(idx / TOTALP);
    int P = (int)(idx % TOTALP);
    int l = 4;
    #pragma unroll
    for (int q = 0; q < 4; ++q) if (P < posBase[q + 1]) { l = q; break; }
    int p  = P - posBase[l];
    int HW = Wl[l] * Wl[l];
    const float* f = (l == 0) ? f0 : (l == 1) ? f1 : (l == 2) ? f2 : (l == 3) ? f3 : f4;
    ft[(size_t)P * 256 + c] = (bf16)f[(size_t)c * HW + p];
}

// =====================================================================
// 3) Fused RPN conv: 3x3/256->256 + bias + ReLU + 1x1 heads (3 cls, 12 box).
//    WG = 512 threads = 16 waves. Each WG: 64 consecutive flat positions,
//    all 256 output channels (one 16-oc tile per wave, 4 pos-tiles of 16).
//    K accumulation: 8 ic-groups x 9 taps of v_wmma_f32_16x16x32_bf16.
//    Staging uses CDNA5 GLOBAL_LOAD_ASYNC_TO_LDS_B128 (ASYNCcnt-tracked)
//    so feature bytes go memory->LDS without a VGPR round trip.
//    SAME-padding handled branchlessly: invalid taps read a zeroed LDS row.
//    B tiles are explicitly load-ahead scheduled (distinct SSA values) so
//    the ds_load pairs overlap WMMA issue instead of s_wait_dscnt 0 stalls.
// =====================================================================
#define ZROW 578
__global__ __launch_bounds__(512)
void k_conv(const bf16* __restrict__ ftL, const bf16* __restrict__ wpack,
            const float* __restrict__ b0, const float* __restrict__ wcls,
            const float* __restrict__ bcls, const float* __restrict__ wbox,
            const float* __restrict__ bbox, float* __restrict__ scores,
            float* __restrict__ deltas, int W, int HW, int posBase) {
    __shared__ __align__(64) bf16 ftile[579][32];   // rows 0..577: halo strip; row 578: zeros
    __shared__ float outacc[15][64];                // fused 1x1 head accumulators
    __shared__ float w1x1[15 * 256];                // cls(3)+box(12) weights

    const int tid  = threadIdx.x;
    const int lane = tid & 31;
    const int wv   = tid >> 5;                      // 0..15: out-channel tile
    const int half = lane >> 4;                     // 0/1: WMMA lane half
    const int n    = lane & 15;                     // N column (position within tile)
    const int pbase = blockIdx.x * 64;
    const int E = 2 * W + 66;

    for (int i = tid; i < 15 * 256; i += 512)
        w1x1[i] = (i < 768) ? wcls[i] : wbox[i - 768];
    for (int i = tid; i < 15 * 64; i += 512)
        ((float*)outacc)[i] = 0.0f;
    if (tid < 16) ((unsigned*)&ftile[ZROW][0])[tid] = 0u;   // zero row for invalid taps

    // per-lane position coordinates + per-(tap,pt) validity bitmask (kk-invariant)
    int px[4], py[4];
    #pragma unroll
    for (int pt = 0; pt < 4; ++pt) {
        int p = pbase + pt * 16 + n;
        py[pt] = p / W;
        px[pt] = p % W;
    }
    unsigned vmask = 0;
    #pragma unroll
    for (int tap = 0; tap < 9; ++tap) {
        const int dy = tap / 3 - 1, dx = tap % 3 - 1;
        #pragma unroll
        for (int pt = 0; pt < 4; ++pt) {
            int sx = px[pt] + dx, sy = py[pt] + dy;
            if ((sx >= 0) & (sx < W) & (sy >= 0) & (sy < W))  // H==W
                vmask |= 1u << (tap * 4 + pt);
        }
    }
    int tapOff[9];
    #pragma unroll
    for (int tap = 0; tap < 9; ++tap)
        tapOff[tap] = (tap / 3) * W + (tap % 3);              // (dy+1)*W + (dx+1)

    v8f acc[4] = {};

    for (int kk = 0; kk < 8; ++kk) {
        __syncthreads();   // previous iteration's B reads done before restaging
        // stage flat halo strip [pbase-W-1, pbase+64+W] x 32 channels, async -> LDS
        for (int i = tid; i < E; i += 512) {
            int sp = pbase - W - 1 + i;
            sp = sp < 0 ? 0 : (sp >= HW ? HW - 1 : sp);
            unsigned lds = (unsigned)(size_t)&ftile[i][0];
            unsigned long long ga =
                (unsigned long long)(size_t)(ftL + (size_t)sp * 256 + kk * 32);
            asm volatile(
                "global_load_async_to_lds_b128 %0, %1, off\n\t"
                "global_load_async_to_lds_b128 %0, %1, off offset:16\n\t"
                "global_load_async_to_lds_b128 %0, %1, off offset:32\n\t"
                "global_load_async_to_lds_b128 %0, %1, off offset:48"
                :: "v"(lds), "v"(ga) : "memory");
        }
        asm volatile("s_wait_asynccnt 0x0" ::: "memory");     // my async writes landed
        __syncthreads();                                       // ... visible to all waves

        #pragma unroll
        for (int tap = 0; tap < 9; ++tap) {
            // A tile: packed weights, one 32B contiguous read per lane
            v16bf amat = *(const v16bf*)(wpack +
                (size_t)(((wv * 9 + tap) * 8 + kk) * 32 + lane) * 16);
            // B tiles: unconditional LDS loads; invalid taps hit the zero row.
            // Explicit load-ahead schedule: keep >=1 ds_load pair in flight
            // ahead of each WMMA so the LDS latency overlaps XDL issue.
            int tl0 = ((vmask >> (tap * 4 + 0)) & 1u) ? (0 * 16 + n + tapOff[tap]) : ZROW;
            int tl1 = ((vmask >> (tap * 4 + 1)) & 1u) ? (1 * 16 + n + tapOff[tap]) : ZROW;
            int tl2 = ((vmask >> (tap * 4 + 2)) & 1u) ? (2 * 16 + n + tapOff[tap]) : ZROW;
            int tl3 = ((vmask >> (tap * 4 + 3)) & 1u) ? (3 * 16 + n + tapOff[tap]) : ZROW;
            v16bf bm0 = *(const v16bf*)&ftile[tl0][lane & 16];
            v16bf bm1 = *(const v16bf*)&ftile[tl1][lane & 16];
            acc[0] = __builtin_amdgcn_wmma_f32_16x16x32_bf16(
                false, amat, false, bm0, (short)0, acc[0], false, false);
            v16bf bm2 = *(const v16bf*)&ftile[tl2][lane & 16];
            acc[1] = __builtin_amdgcn_wmma_f32_16x16x32_bf16(
                false, amat, false, bm1, (short)0, acc[1], false, false);
            v16bf bm3 = *(const v16bf*)&ftile[tl3][lane & 16];
            acc[2] = __builtin_amdgcn_wmma_f32_16x16x32_bf16(
                false, amat, false, bm2, (short)0, acc[2], false, false);
            acc[3] = __builtin_amdgcn_wmma_f32_16x16x32_bf16(
                false, amat, false, bm3, (short)0, acc[3], false, false);
        }
    }

    // epilogue: bias + ReLU, then fused 1x1 heads reduced via LDS f32 atomics
    #pragma unroll
    for (int pt = 0; pt < 4; ++pt) {
        float h[8];
        #pragma unroll
        for (int v = 0; v < 8; ++v) {
            int oc = wv * 16 + half * 8 + v;
            float hv = acc[pt][v] + b0[oc];
            h[v] = hv > 0.0f ? hv : 0.0f;
        }
        int col = pt * 16 + n;
        #pragma unroll
        for (int a = 0; a < 15; ++a) {
            float s = 0.0f;
            #pragma unroll
            for (int v = 0; v < 8; ++v)
                s += h[v] * w1x1[a * 256 + wv * 16 + half * 8 + v];
            atomicAdd(&outacc[a][col], s);   // ds_add_f32
        }
    }
    __syncthreads();

    for (int i = tid; i < 960; i += 512) {
        int a = i >> 6, q = i & 63;
        int p = pbase + q;
        float val = outacc[a][q] + (a < 3 ? bcls[a] : bbox[a - 3]);
        int P3 = (posBase + p) * 3;
        if (a < 3) {
            scores[P3 + a] = val;
        } else {
            int c = a - 3;
            deltas[(size_t)(P3 + (c >> 2)) * 4 + (c & 3)] = val;
        }
    }
}

// ---------------- monotone float->uint key ----------------
__device__ __forceinline__ unsigned fkey(float s) {
    unsigned u = __float_as_uint(s);
    return (u & 0x80000000u) ? ~u : (u | 0x80000000u);
}

// =====================================================================
// 4) Radix-select the 6000th largest score key (4 x 8-bit passes, 1 WG).
// =====================================================================
__global__ __launch_bounds__(1024)
void k_select(const float* __restrict__ scores, unsigned* __restrict__ outT) {
    __shared__ unsigned hist[256];
    __shared__ unsigned prefixS, maskS, needS;
    int tid = threadIdx.x;
    if (tid == 0) { prefixS = 0; maskS = 0; needS = KPRE; }
    __syncthreads();
    for (int pass = 0; pass < 4; ++pass) {
        int shift = 24 - 8 * pass;
        if (tid < 256) hist[tid] = 0;
        __syncthreads();
        unsigned pm = maskS, pv = prefixS;
        for (int i = tid; i < TOTALA; i += 1024) {
            unsigned key = fkey(scores[i]);
            if ((key & pm) == pv) atomicAdd(&hist[(key >> shift) & 255], 1u);
        }
        __syncthreads();
        if (tid == 0) {
            unsigned c = 0, need = needS;
            for (int b = 255; b >= 0; --b) {
                if (c + hist[b] >= need) {
                    prefixS = pv | ((unsigned)b << shift);
                    maskS   = pm | (0xFFu << shift);
                    needS   = need - c;
                    break;
                }
                c += hist[b];
            }
        }
        __syncthreads();
    }
    if (tid == 0) *outT = prefixS;
}

__global__ void k_init(unsigned long long* selk, unsigned* counter) {
    int t = blockIdx.x * blockDim.x + threadIdx.x;
    if (t < CAP) selk[t] = 0ull;
    if (t == 0) *counter = 0u;
}

__global__ void k_gather(const float* __restrict__ scores, const unsigned* __restrict__ thrP,
                         unsigned long long* __restrict__ selk, unsigned* __restrict__ counter) {
    int i = blockIdx.x * blockDim.x + threadIdx.x;
    if (i >= TOTALA) return;
    unsigned key = fkey(scores[i]);
    if (key >= *thrP) {
        unsigned pos = atomicAdd(counter, 1u);
        if (pos < CAP)
            selk[pos] = ((unsigned long long)key << 32) | (unsigned)(~(unsigned)i);
    }
}

// =====================================================================
// 5) Bitonic sort of 8192 u64 keys (descending) in 64KB dynamic LDS.
//    Tie-break via ~index keeps lax.top_k's lower-index-first ordering.
// =====================================================================
__global__ __launch_bounds__(1024)
void k_sort(unsigned long long* __restrict__ selk, unsigned* __restrict__ order) {
    extern __shared__ unsigned long long sk[];
    int tid = threadIdx.x;
    for (int i = tid; i < CAP; i += 1024) sk[i] = selk[i];
    __syncthreads();
    for (int k2 = 2; k2 <= CAP; k2 <<= 1) {
        for (int j = k2 >> 1; j > 0; j >>= 1) {
            for (int i = tid; i < CAP; i += 1024) {
                int l = i ^ j;
                if (l > i) {
                    unsigned long long a = sk[i], b = sk[l];
                    bool sw = ((i & k2) == 0) ? (a < b) : (a > b);  // descending
                    if (sw) { sk[i] = b; sk[l] = a; }
                }
            }
            __syncthreads();
        }
    }
    for (int j = tid; j < KPRE; j += 1024)
        order[j] = ~((unsigned)(sk[j] & 0xFFFFFFFFull));
}

// =====================================================================
// 6) Decode: regenerate anchor on the fly, apply deltas, clip to image.
// =====================================================================
__global__ void k_decode(const unsigned* __restrict__ order, const float* __restrict__ deltas,
                         float* __restrict__ boxes) {
    int j = blockIdx.x * blockDim.x + threadIdx.x;
    if (j >= KPRE) return;
    unsigned i = order[j];
    if (i >= (unsigned)TOTALA) {
        boxes[j * 4 + 0] = 0.f; boxes[j * 4 + 1] = 0.f;
        boxes[j * 4 + 2] = 0.f; boxes[j * 4 + 3] = 0.f;
        return;
    }
    const int aBase[6] = {0, 196608, 245760, 258048, 261120, 261888};
    const int Wl[5]    = {256, 128, 64, 32, 16};
    const int Sl[5]    = {4, 8, 16, 32, 64};
    int l = 4;
    #pragma unroll
    for (int q = 0; q < 4; ++q) if ((int)i < aBase[q + 1]) { l = q; break; }
    int rem = (int)i - aBase[l];
    int p = rem / 3, a = rem % 3;
    int W = Wl[l], s = Sl[l];
    int y = p / W, x = p % W;
    float ratio = (a == 0) ? 0.5f : ((a == 1) ? 1.0f : 2.0f);
    float wpre = rintf(sqrtf((float)(s * s) / ratio));   // rintf == numpy round-half-even
    float hpre = rintf(wpre * ratio);
    float wa = wpre * 8.0f, ha = hpre * 8.0f;            // scale/stride == 8 all levels
    float cx = (float)x * s + (s - 1) * 0.5f;
    float cy = (float)y * s + (s - 1) * 0.5f;
    float ax1 = cx - 0.5f * (wa - 1.0f);
    float ay1 = cy - 0.5f * (ha - 1.0f);
    float ax2 = cx + 0.5f * (wa - 1.0f) + 1.0f;
    float ay2 = cy + 0.5f * (ha - 1.0f) + 1.0f;
    float wA = ax2 - ax1, hA = ay2 - ay1;
    float xa = 0.5f * (ax1 + ax2), ya = 0.5f * (ay1 + ay2);
    float tx = deltas[(size_t)i * 4 + 0], ty = deltas[(size_t)i * 4 + 1];
    float tw = deltas[(size_t)i * 4 + 2], th = deltas[(size_t)i * 4 + 3];
    const float clipv = 4.4248627056f;                   // log(1333/16)
    float wb = expf(fminf(tw, clipv)) * wA;
    float hb = expf(fminf(th, clipv)) * hA;
    float xb = tx * wA + xa, yb = ty * hA + ya;
    boxes[j * 4 + 0] = fminf(fmaxf(xb - 0.5f * wb, 0.f), 1024.f);
    boxes[j * 4 + 1] = fminf(fmaxf(yb - 0.5f * hb, 0.f), 1024.f);
    boxes[j * 4 + 2] = fminf(fmaxf(xb + 0.5f * wb, 0.f), 1024.f);
    boxes[j * 4 + 3] = fminf(fmaxf(yb + 0.5f * hb, 0.f), 1024.f);
}

// =====================================================================
// 7) Suppression-bitmask matrix: mask[i][w] bit b set iff IoU(i, w*32+b)>0.7
//    (only j>i bits used by the scan).
// =====================================================================
__global__ __launch_bounds__(192)
void k_mask(const float* __restrict__ boxes, unsigned* __restrict__ mask) {
    int i = blockIdx.x;
    int w = threadIdx.x;
    if (w >= NWORDS) return;
    float x1 = boxes[i * 4 + 0], y1 = boxes[i * 4 + 1];
    float x2 = boxes[i * 4 + 2], y2 = boxes[i * 4 + 3];
    float ai = (x2 - x1) * (y2 - y1);
    unsigned bits = 0;
    #pragma unroll 4
    for (int b = 0; b < 32; ++b) {
        int j = w * 32 + b;
        if (j > i && j < KPRE) {
            float X1 = boxes[j * 4 + 0], Y1 = boxes[j * 4 + 1];
            float X2 = boxes[j * 4 + 2], Y2 = boxes[j * 4 + 3];
            float aj = (X2 - X1) * (Y2 - Y1);
            float iw = fminf(x2, X2) - fmaxf(x1, X1); iw = iw > 0.f ? iw : 0.f;
            float ih = fminf(y2, Y2) - fmaxf(y1, Y1); ih = ih > 0.f ? ih : 0.f;
            float inter = iw * ih;
            float iou = inter / (ai + aj - inter);
            if (iou > 0.7f) bits |= 1u << b;
        }
    }
    mask[(size_t)i * NWORDS + w] = bits;
}

// =====================================================================
// 8) Greedy NMS scan. Single wave32 workgroup: the per-iteration
//    __syncthreads() degenerates to S_NOP (no barrier resource for a
//    1-wave WG) but fences the LDS accesses for the compiler.
// =====================================================================
__global__ void k_nms(const unsigned* __restrict__ mask, unsigned* __restrict__ keep) {
    __shared__ unsigned remv[NWORDS];
    int lane = threadIdx.x;            // 32 threads
    for (int w = lane; w < NWORDS; w += 32) remv[w] = 0;
    __syncthreads();
    for (int i = 0; i < KPRE; ++i) {
        unsigned bit = (remv[i >> 5] >> (i & 31)) & 1u;
        if (!bit) {
            for (int w = lane; w < NWORDS; w += 32)
                remv[w] |= mask[(size_t)i * NWORDS + w];
        }
        if (lane == 0) keep[i] = bit ^ 1u;
        __syncthreads();
    }
}

// =====================================================================
// 9) Finalize: stable prefix scan of keep == argsort(~keep)[:1000] order;
//    zero-fill then scatter kept boxes by rank.
// =====================================================================
__global__ __launch_bounds__(1024)
void k_final(const unsigned* __restrict__ keep, const float* __restrict__ boxes,
             float* __restrict__ out) {
    __shared__ unsigned arr[1024];
    __shared__ unsigned totalS;
    int tid = threadIdx.x;
    for (int t = tid; t < KPOST * 4; t += 1024) out[t] = 0.0f;
    if (tid == 0) totalS = 0;
    __syncthreads();
    for (int chunk = 0; chunk < 6; ++chunk) {
        int idx = chunk * 1024 + tid;
        unsigned v = (idx < KPRE) ? keep[idx] : 0u;
        arr[tid] = v;
        __syncthreads();
        for (int off = 1; off < 1024; off <<= 1) {
            unsigned t2 = (tid >= off) ? arr[tid - off] : 0u;
            __syncthreads();
            arr[tid] += t2;
            __syncthreads();
        }
        unsigned rank = totalS + arr[tid] - v;
        if (idx < KPRE && v && rank < KPOST) {
            out[rank * 4 + 0] = boxes[idx * 4 + 0];
            out[rank * 4 + 1] = boxes[idx * 4 + 1];
            out[rank * 4 + 2] = boxes[idx * 4 + 2];
            out[rank * 4 + 3] = boxes[idx * 4 + 3];
        }
        __syncthreads();
        if (tid == 0) totalS += arr[1023];
        __syncthreads();
    }
}

// =====================================================================
// host-side launch
// =====================================================================
extern "C" void kernel_launch(void* const* d_in, const int* in_sizes, int n_in,
                              void* d_out, int out_size, void* d_ws, size_t ws_size,
                              hipStream_t stream) {
    if (ws_size < WS_NEEDED) return;  // workspace too small: cannot run

    const float* f0   = (const float*)d_in[1];
    const float* f1   = (const float*)d_in[2];
    const float* f2   = (const float*)d_in[3];
    const float* f3   = (const float*)d_in[4];
    const float* f4   = (const float*)d_in[5];
    const float* w0   = (const float*)d_in[6];
    const float* b0   = (const float*)d_in[7];
    const float* wcls = (const float*)d_in[8];
    const float* bcls = (const float*)d_in[9];
    const float* wbox = (const float*)d_in[10];
    const float* bbox = (const float*)d_in[11];

    char* ws = (char*)d_ws;
    bf16*               ft      = (bf16*)(ws + OFF_FT);
    bf16*               wpack   = (bf16*)(ws + OFF_WPACK);
    float*              scores  = (float*)(ws + OFF_SCORES);
    float*              deltas  = (float*)(ws + OFF_DELTAS);
    unsigned long long* selk    = (unsigned long long*)(ws + OFF_SELK);
    unsigned*           order   = (unsigned*)(ws + OFF_ORDER);
    float*              boxes   = (float*)(ws + OFF_BOXES);
    unsigned*           mask    = (unsigned*)(ws + OFF_MASK);
    unsigned*           keep    = (unsigned*)(ws + OFF_KEEP);
    unsigned*           misc    = (unsigned*)(ws + OFF_MISC);
    unsigned*           thr     = misc;
    unsigned*           counter = misc + 1;

    k_pack_w<<<144, 256, 0, stream>>>(w0, wpack);
    {
        long long tot = (long long)TOTALP * 256;
        int blocks = (int)((tot + 255) / 256);
        k_transpose<<<blocks, 256, 0, stream>>>(f0, f1, f2, f3, f4, ft);
    }

    const int Wl[5]   = {256, 128, 64, 32, 16};
    const int posB[5] = {0, 65536, 81920, 86016, 87040};
    for (int l = 0; l < 5; ++l) {
        int HW = Wl[l] * Wl[l];
        k_conv<<<HW / 64, 512, 0, stream>>>(ft + (size_t)posB[l] * 256, wpack,
                                            b0, wcls, bcls, wbox, bbox,
                                            scores, deltas, Wl[l], HW, posB[l]);
    }

    k_select<<<1, 1024, 0, stream>>>(scores, thr);
    k_init<<<(CAP + 255) / 256, 256, 0, stream>>>(selk, counter);
    k_gather<<<(TOTALA + 255) / 256, 256, 0, stream>>>(scores, thr, selk, counter);
    k_sort<<<1, 1024, CAP * sizeof(unsigned long long), stream>>>(selk, order);
    k_decode<<<(KPRE + 255) / 256, 256, 0, stream>>>(order, deltas, boxes);
    k_mask<<<KPRE, 192, 0, stream>>>(boxes, mask);
    k_nms<<<1, 32, 0, stream>>>(mask, keep);
    k_final<<<1, 1024, 0, stream>>>(keep, boxes, (float*)d_out);
}
